// NTXentLoss_51067161149841
// MI455X (gfx1250) — compile-verified, matrix operand order checked by reading, not run on hardware
//
#include <hip/hip_runtime.h>
#include <hip/hip_bf16.h>

// NT-Xent loss, fused: normalize -> f16, then WMMA X*X^T with streaming
// fixed-max logsumexp (sim/T bounded by 2.0), never materializing sim.
// v2: 32 rows per wave (two independent WMMA chains per B fragment) +
//     explicit B double-buffering so loads overlap the matrix pipe.

typedef __attribute__((ext_vector_type(16))) _Float16 v16h;
typedef __attribute__((ext_vector_type(8)))  float    v8f;

#define B_ROWS 4096
#define N_ROWS 8192
#define DIMS   256
#define NTILES (N_ROWS / 16)
#define NEG_BIG (-1.0e9f)

// ---------------- Phase 0: zero the scalar output ----------------
__global__ void ntx_zero(float* out) {
    if (threadIdx.x == 0 && blockIdx.x == 0) out[0] = 0.0f;
}

// ---------------- Phase 1: L2-normalize rows, emit f16 ----------------
__global__ void ntx_normalize(const float* __restrict__ zi,
                              const float* __restrict__ zj,
                              _Float16* __restrict__ zn) {
    const int row  = (blockIdx.x * blockDim.x + threadIdx.x) >> 5;
    const int lane = threadIdx.x & 31;
    if (row >= N_ROWS) return;
    const float* src = (row < B_ROWS) ? (zi + (size_t)row * DIMS)
                                      : (zj + (size_t)(row - B_ROWS) * DIMS);
    float v[8];
    float ss = 0.0f;
#pragma unroll
    for (int i = 0; i < 8; ++i) {
        v[i] = src[lane + 32 * i];
        ss = fmaf(v[i], v[i], ss);
    }
#pragma unroll
    for (int m = 16; m >= 1; m >>= 1) ss += __shfl_xor(ss, m, 32);
    const float scale = 1.0f / fmaxf(sqrtf(ss), 1e-8f);
    _Float16* dst = zn + (size_t)row * DIMS;
#pragma unroll
    for (int i = 0; i < 8; ++i) dst[lane + 32 * i] = (_Float16)(v[i] * scale);
}

// ---------------- Phase 2: fused WMMA GEMM + logsumexp ----------------
// Fragment layout (16-bit 16x32): lane L<16 holds row L, K = {kb..kb+7, kb+16..kb+23};
// lanes 16..31 hold row L-16 with K offset +8. For X*X^T the B tile loads
// identically (lane = column index = zn row). C layout: lane = N, VGPR = M
// (M += 8 for the hi 16 lanes).

__device__ __forceinline__ void ntx_load_frag8(v16h* F, const _Float16* base) {
#pragma unroll
    for (int c = 0; c < 8; ++c) {
        *(uint4*)&F[c]       = *(const uint4*)(base + c * 32);
        *((uint4*)&F[c] + 1) = *(const uint4*)(base + c * 32 + 16);
    }
}

__device__ __forceinline__ void ntx_epilogue(v8f acc, int jt, int it, int itP,
                                             int dv, v8f& s, float& pos) {
    if (jt == it) {            // mask self-similarity on this tile's diagonal
#pragma unroll
        for (int v = 0; v < 8; ++v)
            if (dv == v) acc[v] = NEG_BIG;
    }
    if (jt == itP) {           // capture positive-pair logit (sim = 2*dot)
#pragma unroll
        for (int v = 0; v < 8; ++v)
            if (dv == v) pos = 2.0f * acc[v];
    }
    // fixed-max streaming logsumexp: logits bounded above by 2.0
#pragma unroll
    for (int v = 0; v < 8; ++v)
        s[v] += __expf(fmaf(2.0f, acc[v], -2.0f));
}

__global__ __launch_bounds__(128)
void ntx_wmma(const _Float16* __restrict__ zn, float* __restrict__ out) {
    const int lane    = threadIdx.x & 31;
    const int wv      = threadIdx.x >> 5;
    const int wpb     = blockDim.x >> 5;
    const int wid     = blockIdx.x * wpb + wv;
    const int nwaves  = gridDim.x * wpb;
    const int halfSel = lane >> 4;
    const int l15     = lane & 15;
    // VGPR index v with (row M = v + 8*halfSel) == (col N = l15); -1 if none
    const int dv = (halfSel == 0) ? ((l15 < 8) ? l15 : -1)
                                  : ((l15 >= 8) ? (l15 - 8) : -1);

    // lane-dependent part of every fragment base address
    const _Float16* fbase = zn + (size_t)l15 * DIMS + halfSel * 8;

    for (int tp = wid; tp < NTILES / 2; tp += nwaves) {
        const int it0 = 2 * tp;
        const int it1 = 2 * tp + 1;
        // Preload both i-tiles' A fragments (rows it0*16.. and it1*16..).
        v16h A0[8], A1[8];
        ntx_load_frag8(A0, fbase + (size_t)it0 * 16 * DIMS);
        ntx_load_frag8(A1, fbase + (size_t)it1 * 16 * DIMS);
        const int itP0 = (it0 < NTILES / 2) ? (it0 + NTILES / 2) : (it0 - NTILES / 2);
        const int itP1 = (it1 < NTILES / 2) ? (it1 + NTILES / 2) : (it1 - NTILES / 2);

        v8f  s0 = {}, s1 = {};
        float pos0 = 0.0f, pos1 = 0.0f;

        // Double-buffered j-tile stream.
        v16h Ba[8], Bb[8];
        ntx_load_frag8(Ba, fbase);             // jt = 0
        for (int jt = 0; jt < NTILES; jt += 2) {
            // prefetch far-ahead tile toward the WGP (speculative, OOB-safe)
            __builtin_prefetch(fbase + (size_t)(jt + 6) * 16 * DIMS, 0, 0);
            // issue loads for jt+1 while computing jt
            ntx_load_frag8(Bb, fbase + (size_t)(jt + 1) * 16 * DIMS);
            {
                v8f acc0 = {}, acc1 = {};
#pragma unroll
                for (int c = 0; c < 8; ++c) {
                    acc0 = __builtin_amdgcn_wmma_f32_16x16x32_f16(
                        false, A0[c], false, Ba[c], (short)0, acc0, false, false);
                    acc1 = __builtin_amdgcn_wmma_f32_16x16x32_f16(
                        false, A1[c], false, Ba[c], (short)0, acc1, false, false);
                }
                ntx_epilogue(acc0, jt, it0, itP0, dv, s0, pos0);
                ntx_epilogue(acc1, jt, it1, itP1, dv, s1, pos1);
            }
            // issue loads for jt+2 (clamped; tail reload is harmless) while computing jt+1
            const int jn = (jt + 2 < NTILES) ? (jt + 2) : 0;
            ntx_load_frag8(Ba, fbase + (size_t)jn * 16 * DIMS);
            {
                v8f acc0 = {}, acc1 = {};
#pragma unroll
                for (int c = 0; c < 8; ++c) {
                    acc0 = __builtin_amdgcn_wmma_f32_16x16x32_f16(
                        false, A0[c], false, Bb[c], (short)0, acc0, false, false);
                    acc1 = __builtin_amdgcn_wmma_f32_16x16x32_f16(
                        false, A1[c], false, Bb[c], (short)0, acc1, false, false);
                }
                ntx_epilogue(acc0, jt + 1, it0, itP0, dv, s0, pos0);
                ntx_epilogue(acc1, jt + 1, it1, itP1, dv, s1, pos1);
            }
        }

        // Sum the 16 column-phases within each 16-lane half.
#pragma unroll
        for (int m = 8; m >= 1; m >>= 1) {
#pragma unroll
            for (int v = 0; v < 8; ++v) {
                s0[v] += __shfl_xor(s0[v], m, 32);
                s1[v] += __shfl_xor(s1[v], m, 32);
            }
        }
        // Diagonal-owning lanes finish their rows: lse - pos (one row per tile).
        float local = 0.0f;
#pragma unroll
        for (int v = 0; v < 8; ++v) {
            if (dv == v) {
                local  = (2.0f + __logf(s0[v])) - pos0;
                local += (2.0f + __logf(s1[v])) - pos1;
            }
        }
#pragma unroll
        for (int m = 16; m >= 1; m >>= 1) local += __shfl_xor(local, m, 32);
        if (lane == 0) atomicAdd(out, local * (1.0f / (float)N_ROWS));
    }
}

extern "C" void kernel_launch(void* const* d_in, const int* in_sizes, int n_in,
                              void* d_out, int out_size, void* d_ws, size_t ws_size,
                              hipStream_t stream) {
    const float* zi = (const float*)d_in[0];
    const float* zj = (const float*)d_in[1];
    float* out      = (float*)d_out;
    _Float16* zn    = (_Float16*)d_ws;   // 8192*256*2 = 4 MB scratch

    ntx_zero<<<1, 32, 0, stream>>>(out);
    // 8192 rows, one wave each, 8 waves/block -> 1024 blocks
    ntx_normalize<<<(N_ROWS * 32) / 256, 256, 0, stream>>>(zi, zj, zn);
    // 256 wave-tasks (32 rows each); 64 blocks x 4 waves = 256 waves
    ntx_wmma<<<64, 128, 0, stream>>>(zn, out);
}
